// GraphRec_21354577396102
// MI455X (gfx1250) — matrix operand order repaired and checked by reading, not compiled, not run on hardware
//
#include <hip/hip_runtime.h>
#include <hip/hip_bf16.h>

typedef __attribute__((ext_vector_type(16))) _Float16 v16h;
typedef __attribute__((ext_vector_type(8)))  _Float16 v8h;
typedef __attribute__((ext_vector_type(8)))  float    v8f;

#define U_N   100000
#define I_N   50000
#define DDIM  64
#define E_UI  1000000
#define E_UU  500000
#define WAVES 4

// ---------------------------------------------------------------------------
// WMMA fragment loaders (CDNA5 wave32 layouts, CDNA5 ISA §7.12.2)
// A 16x32 f16: lane L: row M=L&15, group g=L>>4; a[i<8]=K(8g+i), a[8+i]=K(16+8g+i)
// B 32x16 f16 (packed by pack_weight_kernel): one contiguous 16-half chunk/lane
// C/D 16x16 f32: c[r] -> (M = r + 8*(L>>4), N = L&15)
// ---------------------------------------------------------------------------

__device__ __forceinline__ v16h load_A_h(const _Float16* A, int lda, int lane, int k0) {
    int row = lane & 15, g = lane >> 4;
    const _Float16* p = A + (size_t)row * lda + k0 + 8 * g;
    v8h lo = *(const v8h*)p;
    v8h hi = *(const v8h*)(p + 16);
    v16h a;
#pragma unroll
    for (int i = 0; i < 8; ++i) { a[i] = lo[i]; a[8 + i] = hi[i]; }
    return a;
}

__device__ __forceinline__ v16h load_A_f(const float* A, int lda, int lane, int k0) {
    int row = lane & 15, g = lane >> 4;
    const float* p = A + (size_t)row * lda + k0 + 8 * g;
    v16h a;
#pragma unroll
    for (int i = 0; i < 8; ++i) { a[i] = (_Float16)p[i]; a[8 + i] = (_Float16)p[16 + i]; }
    return a;
}

// packed weights: tile t = kc*4 + nt (Nout fixed = 64), 512 halves per tile,
// element [lane*16 + i] = W[kc*32 + (i<8 ? 8*(lane>>4)+i : 16+8*(lane>>4)+i-8)][nt*16 + (lane&15)]
__device__ __forceinline__ v16h load_B_p(const _Float16* wp, int kc, int nt, int lane) {
    return *(const v16h*)(wp + ((size_t)(kc * 4 + nt) << 9) + lane * 16);
}

// out[16][64] = act(in[16][KIN] @ W + bias), out stored f16 with row stride ldo
template <int KIN>
__device__ __forceinline__ void wave_layer_h(const _Float16* A, int lda,
                                             const _Float16* wp, const float* bias,
                                             _Float16* out, int ldo, float slope, int lane) {
    int col = lane & 15, g = lane >> 4;
#pragma unroll
    for (int nt = 0; nt < 4; ++nt) {
        v8f c = {};
#pragma unroll
        for (int kc = 0; kc < (KIN >> 5); ++kc) {
            v16h a = load_A_h(A, lda, lane, kc * 32);
            v16h b = load_B_p(wp, kc, nt, lane);
            c = __builtin_amdgcn_wmma_f32_16x16x32_f16(false, a, false, b, (short)0, c, false, false);
        }
        float bb = bias[nt * 16 + col];
#pragma unroll
        for (int r = 0; r < 8; ++r) {
            float v = c[r] + bb;
            v = v > 0.f ? v : slope * v;
            out[(size_t)(r + 8 * g) * ldo + nt * 16 + col] = (_Float16)v;
        }
    }
}

// ---------------------------------------------------------------------------
// Weight packing: fp32 [K][64] -> f16 WMMA B-tile layout
// ---------------------------------------------------------------------------
__global__ void pack_weight_kernel(const float* __restrict__ W, _Float16* __restrict__ out,
                                   int K, int N) {
    int d = blockIdx.x * blockDim.x + threadIdx.x;
    if (d >= K * N) return;
    int t = d >> 9, r = d & 511;
    int lane = r >> 4, i = r & 15;
    int Ntiles = N >> 4;
    int kc = t / Ntiles, nt = t % Ntiles;
    int Kidx = kc * 32 + (i < 8 ? 8 * (lane >> 4) + i : 16 + 8 * (lane >> 4) + (i - 8));
    int Nidx = nt * 16 + (lane & 15);
    out[d] = (_Float16)W[(size_t)Kidx * N + Nidx];
}

__global__ void cvt_f32_f16_kernel(const float* __restrict__ in, _Float16* __restrict__ out, int n) {
    int i = blockIdx.x * blockDim.x + threadIdx.x;
    if (i < n) out[i] = (_Float16)in[i];
}

__global__ void fill_f32_kernel(float* __restrict__ p, float v, int n) {
    int i = blockIdx.x * blockDim.x + threadIdx.x;
    if (i < n) p[i] = v;
}

// ---------------------------------------------------------------------------
// Edge feature gather: feat[e][0:64]=a[ai[e]], feat[e][64:128]=b[bi[e]]  (f16)
// ---------------------------------------------------------------------------
__global__ void gather_feat_kernel(const float* __restrict__ a, const int* __restrict__ ai,
                                   const float* __restrict__ b, const int* __restrict__ bi,
                                   _Float16* __restrict__ feat, int nE) {
    int idx = blockIdx.x * blockDim.x + threadIdx.x;
    if (idx >= nE * 128) return;
    int e = idx >> 7, c = idx & 127;
    float v = (c < 64) ? a[(size_t)ai[e] * 64 + c] : b[(size_t)bi[e] * 64 + (c - 64)];
    feat[idx] = (_Float16)v;
}

// ---------------------------------------------------------------------------
// Fused per-edge MLP + attention: one wave = 16 edges.
//   mid = act(act(feat @ W1 + b1) @ W2 + b2)               (slope = mlp_slope)
//   sc  = att3( relu(att2( relu(att1( [mid, dstEmb] )))) )
// ---------------------------------------------------------------------------
__global__ void __launch_bounds__(WAVES * 32)
edge_mlp_att_kernel(const _Float16* __restrict__ feat,
                    const _Float16* wp1, const float* b1,
                    const _Float16* wp2, const float* b2, float mlp_slope,
                    const _Float16* aw1, const float* ab1,
                    const _Float16* aw2, const float* ab2,
                    const _Float16* aw3, const float* ab3,
                    const float* __restrict__ dstEmb, const int* __restrict__ dstIdx,
                    _Float16* __restrict__ outMid, float* __restrict__ outSc, int nEdges) {
    __shared__ __align__(32) _Float16 smem[WAVES * 3072];
    int wid = threadIdx.x >> 5, lane = threadIdx.x & 31;
    int tile = blockIdx.x * WAVES + wid;
    if (tile * 16 >= nEdges) return;
    int e0 = tile * 16;
    _Float16* attIn = smem + wid * 3072;   // 16 x 128
    _Float16* tmp   = attIn + 2048;        // 16 x 64

    // feature MLP layer 1 (128 -> 64), A straight from global f16 feat
    wave_layer_h<128>(feat + (size_t)e0 * 128, 128, wp1, b1, tmp, 64, mlp_slope, lane);
    __builtin_amdgcn_wave_barrier();
    // layer 2 (64 -> 64) -> attIn[:, 0:64]
    wave_layer_h<64>(tmp, 64, wp2, b2, attIn, 128, mlp_slope, lane);
    __builtin_amdgcn_wave_barrier();

    // write mid to global; gather destination embedding into attIn[:, 64:128]
    for (int idx = lane; idx < 16 * 64; idx += 32) {
        int r = idx >> 6, c = idx & 63;
        outMid[(size_t)(e0 + r) * 64 + c] = attIn[r * 128 + c];
    }
    for (int idx = lane; idx < 16 * 64; idx += 32) {
        int r = idx >> 6, c = idx & 63;
        int n = dstIdx[e0 + r];
        attIn[r * 128 + 64 + c] = (_Float16)dstEmb[(size_t)n * 64 + c];
    }
    __builtin_amdgcn_wave_barrier();

    // attention MLP: relu(128->64), relu(64->64), dot with w3
    wave_layer_h<128>(attIn, 128, aw1, ab1, tmp, 64, 0.0f, lane);
    __builtin_amdgcn_wave_barrier();
    wave_layer_h<64>(tmp, 64, aw2, ab2, attIn, 64, 0.0f, lane);
    __builtin_amdgcn_wave_barrier();

    if (lane < 16) {
        float acc = ab3[0];
#pragma unroll
        for (int i = 0; i < 64; ++i)
            acc += (float)attIn[lane * 64 + i] * (float)aw3[i];
        outSc[e0 + lane] = acc;
    }
}

// SocialAgg attention only: attIn = [userEmb[trust], hI[trustee]] -> score
__global__ void __launch_bounds__(WAVES * 32)
social_att_kernel(const float* __restrict__ userEmb, const float* __restrict__ hI,
                  const int* __restrict__ trustE, const int* __restrict__ trusteeE,
                  const _Float16* aw1, const float* ab1,
                  const _Float16* aw2, const float* ab2,
                  const _Float16* aw3, const float* ab3,
                  float* __restrict__ outSc, int nEdges) {
    __shared__ __align__(32) _Float16 smem[WAVES * 3072];
    int wid = threadIdx.x >> 5, lane = threadIdx.x & 31;
    int tile = blockIdx.x * WAVES + wid;
    if (tile * 16 >= nEdges) return;
    int e0 = tile * 16;
    _Float16* attIn = smem + wid * 3072;
    _Float16* tmp   = attIn + 2048;

    for (int idx = lane; idx < 16 * 64; idx += 32) {
        int r = idx >> 6, c = idx & 63;
        attIn[r * 128 + c]      = (_Float16)userEmb[(size_t)trustE[e0 + r] * 64 + c];
        attIn[r * 128 + 64 + c] = (_Float16)hI[(size_t)trusteeE[e0 + r] * 64 + c];
    }
    __builtin_amdgcn_wave_barrier();
    wave_layer_h<128>(attIn, 128, aw1, ab1, tmp, 64, 0.0f, lane);
    __builtin_amdgcn_wave_barrier();
    wave_layer_h<64>(tmp, 64, aw2, ab2, attIn, 64, 0.0f, lane);
    __builtin_amdgcn_wave_barrier();
    if (lane < 16) {
        float acc = ab3[0];
#pragma unroll
        for (int i = 0; i < 64; ++i)
            acc += (float)attIn[lane * 64 + i] * (float)aw3[i];
        outSc[e0 + lane] = acc;
    }
}

// ---------------------------------------------------------------------------
// Edge softmax pieces
// ---------------------------------------------------------------------------
__global__ void seg_max_kernel(const float* __restrict__ sc, const int* __restrict__ seg,
                               float* __restrict__ m, int n) {
    int i = blockIdx.x * blockDim.x + threadIdx.x;
    if (i >= n) return;
    float v = sc[i];
    float* a = &m[seg[i]];
    if (v >= 0.f) atomicMax((int*)a, __float_as_int(v));
    else          atomicMin((unsigned int*)a, __float_as_uint(v));
}

__global__ void seg_expsum_kernel(float* __restrict__ sc, const int* __restrict__ seg,
                                  const float* __restrict__ m, float* __restrict__ s, int n) {
    int i = blockIdx.x * blockDim.x + threadIdx.x;
    if (i >= n) return;
    int g = seg[i];
    float e = __expf(sc[i] - m[g]);
    sc[i] = e;
    atomicAdd(&s[g], e);
}

// acc[seg[e]] += (sc[e]/s[seg[e]]) * mid[e]   (mid f16)
__global__ void scatter_mid_kernel(const _Float16* __restrict__ mid, const float* __restrict__ sc,
                                   const float* __restrict__ s, const int* __restrict__ seg,
                                   float* __restrict__ acc, int nE) {
    int idx = blockIdx.x * blockDim.x + threadIdx.x;
    if (idx >= nE * 64) return;
    int e = idx >> 6, d = idx & 63;
    int g = seg[e];
    float val = sc[e] / s[g];
    atomicAdd(&acc[(size_t)g * 64 + d], (float)mid[(size_t)e * 64 + d] * val);
}

// acc[seg[e]] += (sc[e]/s[seg[e]]) * src[srcIdx[e]]   (src f32)
__global__ void scatter_gather_kernel(const float* __restrict__ src, const int* __restrict__ srcIdx,
                                      const float* __restrict__ sc, const float* __restrict__ s,
                                      const int* __restrict__ seg, float* __restrict__ acc, int nE) {
    int idx = blockIdx.x * blockDim.x + threadIdx.x;
    if (idx >= nE * 64) return;
    int e = idx >> 6, d = idx & 63;
    int g = seg[e];
    float val = sc[e] / s[g];
    atomicAdd(&acc[(size_t)g * 64 + d], src[(size_t)srcIdx[e] * 64 + d] * val);
}

// ---------------------------------------------------------------------------
// Dense node layers (WMMA): out = act(in[rows][64] @ W[64][64] + b)
// ---------------------------------------------------------------------------
__global__ void __launch_bounds__(WAVES * 32)
dense64_kernel(const float* __restrict__ in, const _Float16* __restrict__ wp,
               const float* __restrict__ bias, float* __restrict__ out, int rows, float slope) {
    int wid = threadIdx.x >> 5, lane = threadIdx.x & 31;
    int tile = blockIdx.x * WAVES + wid;
    if (tile * 16 >= rows) return;
    const float* A = in + (size_t)tile * 16 * 64;
    int col = lane & 15, g = lane >> 4;
#pragma unroll
    for (int nt = 0; nt < 4; ++nt) {
        v8f c = {};
#pragma unroll
        for (int kc = 0; kc < 2; ++kc) {
            v16h a = load_A_f(A, 64, lane, kc * 32);
            v16h b = load_B_p(wp, kc, nt, lane);
            c = __builtin_amdgcn_wmma_f32_16x16x32_f16(false, a, false, b, (short)0, c, false, false);
        }
        float bb = bias[nt * 16 + col];
#pragma unroll
        for (int r = 0; r < 8; ++r) {
            float v = c[r] + bb;
            v = v > 0.f ? v : slope * v;
            out[(size_t)(tile * 16 + r + 8 * g) * 64 + nt * 16 + col] = v;
        }
    }
}

// out = act([inA, inB][rows][128] @ W[128][64] + b)
__global__ void __launch_bounds__(WAVES * 32)
dense128cat_kernel(const float* __restrict__ inA, const float* __restrict__ inB,
                   const _Float16* __restrict__ wp, const float* __restrict__ bias,
                   float* __restrict__ out, int rows, float slope) {
    int wid = threadIdx.x >> 5, lane = threadIdx.x & 31;
    int tile = blockIdx.x * WAVES + wid;
    if (tile * 16 >= rows) return;
    size_t base = (size_t)tile * 16 * 64;
    int col = lane & 15, g = lane >> 4;
#pragma unroll
    for (int nt = 0; nt < 4; ++nt) {
        v8f c = {};
#pragma unroll
        for (int kc = 0; kc < 4; ++kc) {
            const float* src = (kc < 2) ? (inA + base) : (inB + base);
            v16h a = load_A_f(src, 64, lane, (kc & 1) * 32);
            v16h b = load_B_p(wp, kc, nt, lane);
            c = __builtin_amdgcn_wmma_f32_16x16x32_f16(false, a, false, b, (short)0, c, false, false);
        }
        float bb = bias[nt * 16 + col];
#pragma unroll
        for (int r = 0; r < 8; ++r) {
            float v = c[r] + bb;
            v = v > 0.f ? v : slope * v;
            out[(size_t)(tile * 16 + r + 8 * g) * 64 + nt * 16 + col] = v;
        }
    }
}

// ---------------------------------------------------------------------------
extern "C" void kernel_launch(void* const* d_in, const int* in_sizes, int n_in,
                              void* d_out, int out_size, void* d_ws, size_t ws_size,
                              hipStream_t stream) {
    (void)in_sizes; (void)n_in; (void)out_size; (void)ws_size;
    const float* user_emb   = (const float*)d_in[0];
    const float* item_emb   = (const float*)d_in[1];
    const float* rating_emb = (const float*)d_in[2];
    const float* gv_w1 = (const float*)d_in[3];  const float* gv_b1 = (const float*)d_in[4];
    const float* gv_w2 = (const float*)d_in[5];  const float* gv_b2 = (const float*)d_in[6];
    const float* attI_w1 = (const float*)d_in[7];  const float* attI_b1 = (const float*)d_in[8];
    const float* attI_w2 = (const float*)d_in[9];  const float* attI_b2 = (const float*)d_in[10];
    const float* attI_w3 = (const float*)d_in[11]; const float* attI_b3 = (const float*)d_in[12];
    const float* wI_w = (const float*)d_in[13];  const float* wI_b = (const float*)d_in[14];
    const float* attS_w1 = (const float*)d_in[15]; const float* attS_b1 = (const float*)d_in[16];
    const float* attS_w2 = (const float*)d_in[17]; const float* attS_b2 = (const float*)d_in[18];
    const float* attS_w3 = (const float*)d_in[19]; const float* attS_b3 = (const float*)d_in[20];
    const float* wS_w = (const float*)d_in[21];  const float* wS_b = (const float*)d_in[22];
    const float* W2_w = (const float*)d_in[23];  const float* W2_b = (const float*)d_in[24];
    const float* gu_w1 = (const float*)d_in[25]; const float* gu_b1 = (const float*)d_in[26];
    const float* gu_w2 = (const float*)d_in[27]; const float* gu_b2 = (const float*)d_in[28];
    const float* attU_w1 = (const float*)d_in[29]; const float* attU_b1 = (const float*)d_in[30];
    const float* attU_w2 = (const float*)d_in[31]; const float* attU_b2 = (const float*)d_in[32];
    const float* attU_w3 = (const float*)d_in[33]; const float* attU_b3 = (const float*)d_in[34];
    const float* wU_w = (const float*)d_in[35];  const float* wU_b = (const float*)d_in[36];
    const int* user_e    = (const int*)d_in[37];
    const int* item_e    = (const int*)d_in[38];
    const int* rating_e  = (const int*)d_in[39];
    const int* trust_e   = (const int*)d_in[40];
    const int* trustee_e = (const int*)d_in[41];

    char* base = (char*)d_ws; size_t off = 0;
    auto alloc = [&](size_t bytes) -> void* {
        void* p = base + off; off = (off + bytes + 255) & ~(size_t)255; return p;
    };
    _Float16* gvW1p = (_Float16*)alloc(128 * 64 * 2);
    _Float16* gvW2p = (_Float16*)alloc(64 * 64 * 2);
    _Float16* aIW1p = (_Float16*)alloc(128 * 64 * 2);
    _Float16* aIW2p = (_Float16*)alloc(64 * 64 * 2);
    _Float16* aIW3p = (_Float16*)alloc(64 * 2);
    _Float16* wIp   = (_Float16*)alloc(64 * 64 * 2);
    _Float16* aSW1p = (_Float16*)alloc(128 * 64 * 2);
    _Float16* aSW2p = (_Float16*)alloc(64 * 64 * 2);
    _Float16* aSW3p = (_Float16*)alloc(64 * 2);
    _Float16* wSp   = (_Float16*)alloc(64 * 64 * 2);
    _Float16* W2p   = (_Float16*)alloc(128 * 64 * 2);
    _Float16* guW1p = (_Float16*)alloc(128 * 64 * 2);
    _Float16* guW2p = (_Float16*)alloc(64 * 64 * 2);
    _Float16* aUW1p = (_Float16*)alloc(128 * 64 * 2);
    _Float16* aUW2p = (_Float16*)alloc(64 * 64 * 2);
    _Float16* aUW3p = (_Float16*)alloc(64 * 2);
    _Float16* wUp   = (_Float16*)alloc(64 * 64 * 2);
    _Float16* feat = (_Float16*)alloc((size_t)E_UI * 128 * 2);
    _Float16* mid  = (_Float16*)alloc((size_t)E_UI * 64 * 2);
    float* sc  = (float*)alloc((size_t)E_UI * 4);
    float* m   = (float*)alloc((size_t)U_N * 4);
    float* s   = (float*)alloc((size_t)U_N * 4);
    float* acc = (float*)alloc((size_t)U_N * 64 * 4);
    float* hI  = (float*)alloc((size_t)U_N * 64 * 4);
    float* hS  = (float*)alloc((size_t)U_N * 64 * 4);

    auto pack = [&](const float* w, _Float16* dst, int K) {
        int n = K * 64;
        pack_weight_kernel<<<(n + 255) / 256, 256, 0, stream>>>(w, dst, K, 64);
    };
    pack(gv_w1, gvW1p, 128);  pack(gv_w2, gvW2p, 64);
    pack(attI_w1, aIW1p, 128); pack(attI_w2, aIW2p, 64);
    pack(wI_w, wIp, 64);
    pack(attS_w1, aSW1p, 128); pack(attS_w2, aSW2p, 64);
    pack(wS_w, wSp, 64);
    pack(W2_w, W2p, 128);
    pack(gu_w1, guW1p, 128);  pack(gu_w2, guW2p, 64);
    pack(attU_w1, aUW1p, 128); pack(attU_w2, aUW2p, 64);
    pack(wU_w, wUp, 64);
    cvt_f32_f16_kernel<<<1, 64, 0, stream>>>(attI_w3, aIW3p, 64);
    cvt_f32_f16_kernel<<<1, 64, 0, stream>>>(attS_w3, aSW3p, 64);
    cvt_f32_f16_kernel<<<1, 64, 0, stream>>>(attU_w3, aUW3p, 64);

    const int tilesUI = E_UI / 16, tilesUU = E_UU / 16;
    const int gEdgeUI = (tilesUI + WAVES - 1) / WAVES;
    const int gEdgeUU = (tilesUU + WAVES - 1) / WAVES;
    const float NEG_INF = -__builtin_huge_valf();

    // ---------------- ItemAgg ----------------
    gather_feat_kernel<<<(E_UI * 128 + 255) / 256, 256, 0, stream>>>(
        item_emb, item_e, rating_emb, rating_e, feat, E_UI);
    edge_mlp_att_kernel<<<gEdgeUI, WAVES * 32, 0, stream>>>(
        feat, gvW1p, gv_b1, gvW2p, gv_b2, 0.01f,
        aIW1p, attI_b1, aIW2p, attI_b2, aIW3p, attI_b3,
        user_emb, user_e, mid, sc, E_UI);
    fill_f32_kernel<<<(U_N + 255) / 256, 256, 0, stream>>>(m, NEG_INF, U_N);
    fill_f32_kernel<<<(U_N + 255) / 256, 256, 0, stream>>>(s, 0.f, U_N);
    seg_max_kernel<<<(E_UI + 255) / 256, 256, 0, stream>>>(sc, user_e, m, E_UI);
    seg_expsum_kernel<<<(E_UI + 255) / 256, 256, 0, stream>>>(sc, user_e, m, s, E_UI);
    fill_f32_kernel<<<(U_N * 64 + 255) / 256, 256, 0, stream>>>(acc, 0.f, U_N * 64);
    scatter_mid_kernel<<<(E_UI * 64 + 255) / 256, 256, 0, stream>>>(mid, sc, s, user_e, acc, E_UI);
    dense64_kernel<<<((U_N / 16) + WAVES - 1) / WAVES, WAVES * 32, 0, stream>>>(
        acc, wIp, wI_b, hI, U_N, 0.2f);

    // ---------------- SocialAgg ----------------
    social_att_kernel<<<gEdgeUU, WAVES * 32, 0, stream>>>(
        user_emb, hI, trust_e, trustee_e,
        aSW1p, attS_b1, aSW2p, attS_b2, aSW3p, attS_b3, sc, E_UU);
    fill_f32_kernel<<<(U_N + 255) / 256, 256, 0, stream>>>(m, NEG_INF, U_N);
    fill_f32_kernel<<<(U_N + 255) / 256, 256, 0, stream>>>(s, 0.f, U_N);
    seg_max_kernel<<<(E_UU + 255) / 256, 256, 0, stream>>>(sc, trustee_e, m, E_UU);
    seg_expsum_kernel<<<(E_UU + 255) / 256, 256, 0, stream>>>(sc, trustee_e, m, s, E_UU);
    fill_f32_kernel<<<(U_N * 64 + 255) / 256, 256, 0, stream>>>(acc, 0.f, U_N * 64);
    scatter_gather_kernel<<<(E_UU * 64 + 255) / 256, 256, 0, stream>>>(
        hI, trust_e, sc, s, trustee_e, acc, E_UU);
    dense64_kernel<<<((U_N / 16) + WAVES - 1) / WAVES, WAVES * 32, 0, stream>>>(
        acc, wSp, wS_b, hS, U_N, 0.2f);
    dense128cat_kernel<<<((U_N / 16) + WAVES - 1) / WAVES, WAVES * 32, 0, stream>>>(
        hI, hS, W2p, W2_b, (float*)d_out, U_N, 0.2f);

    // ---------------- UserAgg ----------------
    gather_feat_kernel<<<(E_UI * 128 + 255) / 256, 256, 0, stream>>>(
        user_emb, user_e, rating_emb, rating_e, feat, E_UI);
    edge_mlp_att_kernel<<<gEdgeUI, WAVES * 32, 0, stream>>>(
        feat, guW1p, gu_b1, guW2p, gu_b2, 0.0f,
        aUW1p, attU_b1, aUW2p, attU_b2, aUW3p, attU_b3,
        item_emb, item_e, mid, sc, E_UI);
    fill_f32_kernel<<<(I_N + 255) / 256, 256, 0, stream>>>(m, NEG_INF, I_N);
    fill_f32_kernel<<<(I_N + 255) / 256, 256, 0, stream>>>(s, 0.f, I_N);
    seg_max_kernel<<<(E_UI + 255) / 256, 256, 0, stream>>>(sc, item_e, m, E_UI);
    seg_expsum_kernel<<<(E_UI + 255) / 256, 256, 0, stream>>>(sc, item_e, m, s, E_UI);
    fill_f32_kernel<<<(I_N * 64 + 255) / 256, 256, 0, stream>>>(acc, 0.f, I_N * 64);
    scatter_mid_kernel<<<(E_UI * 64 + 255) / 256, 256, 0, stream>>>(mid, sc, s, item_e, acc, E_UI);
    dense64_kernel<<<((I_N / 16) + WAVES - 1) / WAVES, WAVES * 32, 0, stream>>>(
        acc, wUp, wU_b, (float*)d_out + (size_t)U_N * 64, I_N, 0.2f);
}
